// GCN_ValueNet_12884901888561
// MI455X (gfx1250) — matrix operand order repaired and checked by reading, not compiled, not run on hardware
//
#include <hip/hip_runtime.h>
#include <hip/hip_bf16.h>
#include <stdint.h>

typedef __attribute__((ext_vector_type(16))) __bf16 v16bf;
typedef __attribute__((ext_vector_type(8)))  __bf16 v8bf;
typedef __attribute__((ext_vector_type(8)))  float  v8f;

// ---------------------------------------------------------------------------
// Degree / normalization
// ---------------------------------------------------------------------------
__global__ void deg_init_k(float* __restrict__ deg, int n) {
    int i = blockIdx.x * blockDim.x + threadIdx.x;
    if (i < n) deg[i] = 1.0f;                 // +1 self loop
}

__global__ void deg_count_k(const long long* __restrict__ dst, float* __restrict__ deg, int E) {
    int e = blockIdx.x * blockDim.x + threadIdx.x;
    if (e < E) atomicAdd(&deg[(int)dst[e]], 1.0f);
}

__global__ void dinv_k(const float* __restrict__ deg, float* __restrict__ dinv, int n) {
    int i = blockIdx.x * blockDim.x + threadIdx.x;
    if (i < n) dinv[i] = rsqrtf(deg[i]);
}

// ---------------------------------------------------------------------------
// Conversions
// ---------------------------------------------------------------------------
__global__ void f32_to_bf16_k(const float* __restrict__ in, __bf16* __restrict__ out, long long n) {
    long long i = (long long)blockIdx.x * blockDim.x + threadIdx.x;
    if (i < n) out[i] = (__bf16)in[i];
}

// W[k*N+n] (row-major K x N, fp32)  ->  Wt[n*K+k] (row-major N x K, bf16)
__global__ void transpose_bf16_k(const float* __restrict__ W, __bf16* __restrict__ Wt, int K, int N) {
    int i = blockIdx.x * blockDim.x + threadIdx.x;
    if (i >= K * N) return;
    int k = i / N, n = i - k * N;
    Wt[(size_t)n * K + k] = (__bf16)W[i];
}

// ---------------------------------------------------------------------------
// Weights-stationary bf16 WMMA GEMM for skinny GEMMs (M huge, K == 128).
// C[M,N] = A[M,128] * B[128,N], B given transposed (Bt[N,128]).
// Each wave owns a 64-column strip: it preloads ALL B fragments for that
// strip (4 k-steps x 4 n-tiles = 128 VGPRs of bf16) into registers, then
// streams `mpw` m-tiles: per m-tile just 8 b128 A-loads and 16 WMMAs.
//
// A fragment (16-bit A 16x32): lane holds A[m, k+half*8+0..7] (elems 0..7)
//                              and  A[m, k+16+half*8+0..7]   (elems 8..15)
// B fragment: lane holds B[k+half*8+0..7, n] (elems 0..7) and B[k+16+half*8+0..7, n] (elems 8..15)
// C/D layout: VGPR r -> row = 16*tm + half*8 + r, col = 16*tn + (lane&15)
// ---------------------------------------------------------------------------
__global__ void __launch_bounds__(256) gemm_bf16_wmma_ws_k(
    const __bf16* __restrict__ A, const __bf16* __restrict__ Bt,
    float* __restrict__ C, int M, int N, int mpw)
{
    constexpr int K = 128;
    const int lane  = threadIdx.x & 31;
    const int wave  = blockIdx.x * (blockDim.x >> 5) + (threadIdx.x >> 5);
    const int strips = N >> 6;                 // 64-column strips
    const int mtiles = M >> 4;
    const int s  = wave % strips;
    const int c  = wave / strips;
    const int m0 = c * mpw;
    if (m0 >= mtiles) return;                  // wave-uniform: EXEC stays all-1s
    const int half = lane >> 4;
    const int l15  = lane & 15;
    const int n0   = s * 64;

    // ---- preload the whole B panel for this strip into registers ----
    v16bf bfrag[4][4];                         // [kstep][ntile]
    #pragma unroll
    for (int ks = 0; ks < 4; ++ks) {
        #pragma unroll
        for (int t = 0; t < 4; ++t) {
            const __bf16* bp = Bt + (size_t)(n0 + t * 16 + l15) * K + ks * 32 + half * 8;
            union { v16bf v; v8bf h[2]; } u;
            u.h[0] = *(const v8bf*)bp;
            u.h[1] = *(const v8bf*)(bp + 16);
            bfrag[ks][t] = u.v;
        }
    }

    const int mend = (m0 + mpw < mtiles) ? (m0 + mpw) : mtiles;
    for (int mt = m0; mt < mend; ++mt) {
        const __bf16* arow = A + (size_t)(mt * 16 + l15) * K;
        v8f acc[4] = {{}, {}, {}, {}};
        #pragma unroll
        for (int ks = 0; ks < 4; ++ks) {
            union { v16bf v; v8bf h[2]; } a;
            a.h[0] = *(const v8bf*)(arow + ks * 32 + half * 8);
            a.h[1] = *(const v8bf*)(arow + ks * 32 + 16 + half * 8);
            #pragma unroll
            for (int t = 0; t < 4; ++t)
                acc[t] = __builtin_amdgcn_wmma_f32_16x16x32_bf16(
                    false, a.v, false, bfrag[ks][t], (short)0, acc[t], false, false);
        }
        float* crow = C + (size_t)(mt * 16 + half * 8) * N + n0 + l15;
        #pragma unroll
        for (int t = 0; t < 4; ++t)
            #pragma unroll
            for (int r = 0; r < 8; ++r)
                crow[(size_t)r * N + t * 16] = acc[t][r];
    }
}

// ---------------------------------------------------------------------------
// GCN aggregation
// ---------------------------------------------------------------------------
// agg[m,f] = h[m,f] * dinv[m]^2 + bias[f]   (self loop + bias folded in)
__global__ void agg_init_k(const float* __restrict__ h, const float* __restrict__ dinv,
                           const float* __restrict__ bias, float* __restrict__ agg,
                           long long total, int F) {
    long long i = (long long)blockIdx.x * blockDim.x + threadIdx.x;
    if (i >= total) return;
    int m = (int)(i / F);
    int f = (int)(i - (long long)m * F);
    float di = dinv[m];
    agg[i] = h[i] * di * di + bias[f];
}

// one wave per edge, F=128: lane owns 4 contiguous floats -> 1x b128 load + 4 atomics
__global__ void edge_scatter128_k(const long long* __restrict__ src, const long long* __restrict__ dst,
                                  const float* __restrict__ h, const float* __restrict__ dinv,
                                  float* __restrict__ agg, int E) {
    int lane = threadIdx.x & 31;
    long long e = ((long long)blockIdx.x * blockDim.x + threadIdx.x) >> 5;
    if (e >= E) return;
    int s = (int)src[e], d = (int)dst[e];
    float norm = dinv[s] * dinv[d];
    const float4 v = *(const float4*)(h + (size_t)s * 128 + lane * 4);
    float* ad = agg + (size_t)d * 128 + lane * 4;
    atomicAdd(ad + 0, v.x * norm);
    atomicAdd(ad + 1, v.y * norm);
    atomicAdd(ad + 2, v.z * norm);
    atomicAdd(ad + 3, v.w * norm);
}

// one wave per edge, F=64: lane owns 2 contiguous floats -> 1x b64 load + 2 atomics
__global__ void edge_scatter64_k(const long long* __restrict__ src, const long long* __restrict__ dst,
                                 const float* __restrict__ h, const float* __restrict__ dinv,
                                 float* __restrict__ agg, int E) {
    int lane = threadIdx.x & 31;
    long long e = ((long long)blockIdx.x * blockDim.x + threadIdx.x) >> 5;
    if (e >= E) return;
    int s = (int)src[e], d = (int)dst[e];
    float norm = dinv[s] * dinv[d];
    const float2 v = *(const float2*)(h + (size_t)s * 64 + lane * 2);
    float* ad = agg + (size_t)d * 64 + lane * 2;
    atomicAdd(ad + 0, v.x * norm);
    atomicAdd(ad + 1, v.y * norm);
}

// scalar (F==1) edge scatter
__global__ void edge_scatter1_k(const long long* __restrict__ src, const long long* __restrict__ dst,
                                const float* __restrict__ h, const float* __restrict__ dinv,
                                float* __restrict__ agg, int E) {
    int e = blockIdx.x * blockDim.x + threadIdx.x;
    if (e >= E) return;
    int s = (int)src[e], d = (int)dst[e];
    atomicAdd(&agg[d], h[s] * dinv[s] * dinv[d]);
}

__global__ void tanh_to_bf16_k(const float* __restrict__ in, __bf16* __restrict__ out, long long n) {
    long long i = (long long)blockIdx.x * blockDim.x + threadIdx.x;
    if (i < n) out[i] = (__bf16)tanhf(in[i]);
}

__global__ void tanh_f32_k(const float* __restrict__ in, float* __restrict__ out, long long n) {
    long long i = (long long)blockIdx.x * blockDim.x + threadIdx.x;
    if (i < n) out[i] = tanhf(in[i]);
}

// ---------------------------------------------------------------------------
// Layer 3 GEMV (K=64, N=1): one wave per row, wave32 shfl reduction
// ---------------------------------------------------------------------------
__global__ void gemv64_bf16_k(const __bf16* __restrict__ H, const float* __restrict__ w,
                              float* __restrict__ out, int n) {
    int lane = threadIdx.x & 31;
    int row  = (blockIdx.x * blockDim.x + threadIdx.x) >> 5;
    if (row >= n) return;
    const __bf16* r = H + (size_t)row * 64;
    float s = (float)r[lane] * w[lane] + (float)r[lane + 32] * w[lane + 32];
    #pragma unroll
    for (int off = 16; off > 0; off >>= 1) s += __shfl_xor(s, off, 32);
    if (lane == 0) out[row] = s;
}

// ---------------------------------------------------------------------------
// Deterministic final reduction: out = dot(W4, h3) + b4
// ---------------------------------------------------------------------------
__global__ void dot_partials_k(const float* __restrict__ W4, const float* __restrict__ h3,
                               float* __restrict__ partials, int n) {
    __shared__ float sm[256];
    int tid = threadIdx.x;
    int i = blockIdx.x * 256 + tid;
    sm[tid] = (i < n) ? W4[i] * h3[i] : 0.0f;
    __syncthreads();
    for (int off = 128; off > 0; off >>= 1) {
        if (tid < off) sm[tid] += sm[tid + off];
        __syncthreads();
    }
    if (tid == 0) partials[blockIdx.x] = sm[0];
}

__global__ void dot_finish_k(const float* __restrict__ partials, int nblk,
                             const float* __restrict__ b4, float* __restrict__ out) {
    __shared__ float sm[256];
    int tid = threadIdx.x;
    float s = 0.0f;
    for (int i = tid; i < nblk; i += 256) s += partials[i];
    sm[tid] = s;
    __syncthreads();
    for (int off = 128; off > 0; off >>= 1) {
        if (tid < off) sm[tid] += sm[tid + off];
        __syncthreads();
    }
    if (tid == 0) out[0] = sm[0] + b4[0];
}

// ---------------------------------------------------------------------------
// Launch
// ---------------------------------------------------------------------------
extern "C" void kernel_launch(void* const* d_in, const int* in_sizes, int n_in,
                              void* d_out, int out_size, void* d_ws, size_t ws_size,
                              hipStream_t stream) {
    const float*     x   = (const float*)d_in[0];
    const long long* ei  = (const long long*)d_in[1];
    const float*     W1  = (const float*)d_in[2];
    const float*     b1  = (const float*)d_in[3];
    const float*     W2  = (const float*)d_in[4];
    const float*     b2  = (const float*)d_in[5];
    const float*     W3  = (const float*)d_in[6];
    const float*     b3  = (const float*)d_in[7];
    const float*     W4  = (const float*)d_in[8];
    const float*     b4  = (const float*)d_in[9];
    float*           out = (float*)d_out;

    const int F1 = 128, H1 = 128, H2 = 64;
    const int Nn = in_sizes[0] / F1;     // 50000 (= 16 * 3125)
    const int E  = in_sizes[1] / 2;      // 800000
    const long long* srcI = ei;
    const long long* dstI = ei + E;

    // Workspace carve-out (~78 MB total, 256B aligned slices)
    char* p = (char*)d_ws;
    auto alloc = [&](size_t bytes) { void* r = (void*)p; p += (bytes + 255) & ~(size_t)255; return r; };
    __bf16* x_bf   = (__bf16*)alloc((size_t)Nn * F1 * 2);
    __bf16* act_bf = (__bf16*)alloc((size_t)Nn * H1 * 2);
    float*  h_f    = (float*) alloc((size_t)Nn * H1 * 4);
    float*  agg_f  = (float*) alloc((size_t)Nn * H1 * 4);
    __bf16* W1t    = (__bf16*)alloc((size_t)F1 * H1 * 2);
    __bf16* W2t    = (__bf16*)alloc((size_t)H1 * H2 * 2);
    float*  deg    = (float*) alloc((size_t)Nn * 4);
    float*  dinv   = (float*) alloc((size_t)Nn * 4);
    float*  pre3   = (float*) alloc((size_t)Nn * 4);
    float*  agg3   = (float*) alloc((size_t)Nn * 4);
    float*  h3     = (float*) alloc((size_t)Nn * 4);
    float*  parts  = (float*) alloc(1024 * 4);

    const int B = 256;
    auto cdiv = [](long long a, long long b) { return (unsigned)((a + b - 1) / b); };

    // --- degrees & normalization ---
    deg_init_k <<<cdiv(Nn, B), B, 0, stream>>>(deg, Nn);
    deg_count_k<<<cdiv(E, B),  B, 0, stream>>>(dstI, deg, E);
    dinv_k     <<<cdiv(Nn, B), B, 0, stream>>>(deg, dinv, Nn);

    // --- precision prep ---
    f32_to_bf16_k   <<<cdiv((long long)Nn * F1, B), B, 0, stream>>>(x, x_bf, (long long)Nn * F1);
    transpose_bf16_k<<<cdiv(F1 * H1, B), B, 0, stream>>>(W1, W1t, F1, H1);
    transpose_bf16_k<<<cdiv(H1 * H2, B), B, 0, stream>>>(W2, W2t, H1, H2);

    const int mtiles = Nn / 16;          // 3125

    // --- Layer 1: h = x @ W1 ; GCN aggregate ; tanh ---
    {
        const int mpw = 2, strips = H1 / 64;                    // 2 strips
        long long waves = (long long)strips * cdiv(mtiles, mpw);
        gemm_bf16_wmma_ws_k<<<cdiv(waves, 8), B, 0, stream>>>(x_bf, W1t, h_f, Nn, H1, mpw);
        agg_init_k<<<cdiv((long long)Nn * H1, B), B, 0, stream>>>(h_f, dinv, b1, agg_f, (long long)Nn * H1, H1);
        edge_scatter128_k<<<cdiv(E, 8), B, 0, stream>>>(srcI, dstI, h_f, dinv, agg_f, E);
        tanh_to_bf16_k<<<cdiv((long long)Nn * H1, B), B, 0, stream>>>(agg_f, act_bf, (long long)Nn * H1);
    }

    // --- Layer 2: h = act @ W2 ; GCN aggregate ; tanh ---
    {
        const int mpw = 1, strips = H2 / 64;                    // 1 strip
        long long waves = (long long)strips * cdiv(mtiles, mpw);
        gemm_bf16_wmma_ws_k<<<cdiv(waves, 8), B, 0, stream>>>(act_bf, W2t, h_f, Nn, H2, mpw);
        agg_init_k<<<cdiv((long long)Nn * H2, B), B, 0, stream>>>(h_f, dinv, b2, agg_f, (long long)Nn * H2, H2);
        edge_scatter64_k<<<cdiv(E, 8), B, 0, stream>>>(srcI, dstI, h_f, dinv, agg_f, E);
        tanh_to_bf16_k<<<cdiv((long long)Nn * H2, B), B, 0, stream>>>(agg_f, act_bf, (long long)Nn * H2);
    }

    // --- Layer 3 (out dim 1): GEMV ; GCN aggregate ; tanh ---
    gemv64_bf16_k<<<cdiv(Nn, 8), B, 0, stream>>>(act_bf, W3, pre3, Nn);
    agg_init_k<<<cdiv(Nn, B), B, 0, stream>>>(pre3, dinv, b3, agg3, (long long)Nn, 1);
    edge_scatter1_k<<<cdiv(E, B), B, 0, stream>>>(srcI, dstI, pre3, dinv, agg3, E);
    tanh_f32_k<<<cdiv(Nn, B), B, 0, stream>>>(agg3, h3, Nn);

    // --- Final: out = dot(W4, h3) + b4 (deterministic 2-pass reduction) ---
    int nblk = (int)cdiv(Nn, B);
    dot_partials_k<<<nblk, B, 0, stream>>>(W4, h3, parts, Nn);
    dot_finish_k<<<1, B, 0, stream>>>(parts, nblk, b4, out);
}
